// Encoder_48146583388404
// MI455X (gfx1250) — matrix operand order repaired and checked by reading, not compiled, not run on hardware
//
#include <hip/hip_runtime.h>
#include <hip/hip_bf16.h>

// ---------------------------------------------------------------------------
// MI455X (gfx1250) 3-layer LSTM encoder.
//
// bf16 WMMA (v_wmma_f32_16x16x32_bf16) with f32 accumulation; weights
// converted once to bf16 and transposed column-major (K contiguous) so they
// are L2-resident (~44MB < 192MB L2) and B-fragments load as contiguous 16B
// ds_load_b128s; fused GEMM + LSTM-cell epilogue with all four gates owned by
// the same lane so the cell update runs on WMMA accumulator registers.
//
// Tile staging uses the CDNA5 async global->LDS path
// (GLOBAL_LOAD_ASYNC_TO_LDS_B128, ASYNCcnt): no VGPR round trip, DMA of
// chunk k+1 overlaps WMMA of chunk k, synchronized with s_wait_asynccnt
// (in-order completion) + barriers. The builtin takes int4-vector pointers
// (AS1 src, AS3 dst) per the clang diagnostic from round 3.
// ---------------------------------------------------------------------------

typedef __bf16  v16bf __attribute__((ext_vector_type(16)));
typedef float   v8f   __attribute__((ext_vector_type(8)));
typedef unsigned int ui4 __attribute__((ext_vector_type(4)));   // 16B chunk
typedef float   flt4  __attribute__((ext_vector_type(4)));
typedef int     i4v   __attribute__((ext_vector_type(4)));

#if defined(__has_builtin)
#if __has_builtin(__builtin_amdgcn_global_load_async_to_lds_b128) && \
    __has_builtin(__builtin_amdgcn_s_wait_asynccnt)
#define ASYNC_LDS 1
#endif
#endif

// 16-byte global -> LDS copy. Async (ASYNCcnt) when available.
static __device__ __forceinline__ void cp16_g2l(const void* gsrc, void* ldst) {
#ifdef ASYNC_LDS
    __builtin_amdgcn_global_load_async_to_lds_b128(
        (__attribute__((address_space(1))) i4v*)(void*)(gsrc),
        (__attribute__((address_space(3))) i4v*)(ldst), 0, 0);
#else
    *(ui4*)ldst = *(const ui4*)gsrc;
#endif
}
template<int N>
static __device__ __forceinline__ void wait_async() {
#ifdef ASYNC_LDS
    __builtin_amdgcn_s_wait_asynccnt((unsigned short)N);
#endif
}

static __device__ __forceinline__ unsigned short f2bf(float f) {
    unsigned u = __float_as_uint(f);
    u += 0x7FFFu + ((u >> 16) & 1u);      // round-to-nearest-even
    return (unsigned short)(u >> 16);
}
static __device__ __forceinline__ float fsigmoid(float x) {
    return 1.0f / (1.0f + __expf(-x));
}
static __device__ __forceinline__ float ftanh(float x) {
    return 2.0f / (1.0f + __expf(-2.0f * x)) - 1.0f;
}

// ---------------------------------------------------------------------------
// Prep: wt[n*Ktot + k] = bf16( k<Kx ? W[k][n] : U[k-Kx][n] ),  n in [0,4096)
// ---------------------------------------------------------------------------
__global__ __launch_bounds__(256) void transpose_cat_kernel(
    const float* __restrict__ W, const float* __restrict__ Ur,
    int Kx, int Ktot, unsigned short* __restrict__ wt)
{
    size_t idx = (size_t)blockIdx.x * 256 + threadIdx.x;   // grid = Ktot*16
    int n = (int)(idx & 4095);
    int k = (int)(idx >> 12);
    float v = (k < Kx) ? W[(size_t)k * 4096 + n]
                       : Ur[(size_t)(k - Kx) * 4096 + n];
    wt[(size_t)n * Ktot + k] = f2bf(v);
}

__global__ __launch_bounds__(256) void init_state_kernel(
    unsigned short* h1, unsigned short* h2, unsigned short* h3,
    float* c1, float* c2, float* c3)
{
    int i = blockIdx.x * 256 + threadIdx.x;        // grid = 1024 -> 262144
    h1[i] = 0; h2[i] = 0; h3[i] = 0;
    c1[i] = 0.0f; c2[i] = 0.0f; c3[i] = 0.0f;
}

// ---------------------------------------------------------------------------
// Fused  z = [x_t , h] @ [W ; U] + b   ->  LSTM cell update.
// Tile: 64 batch x 64 u per WG (x4 gates). 8 waves: 4(M) x 2(U); each wave
// one 16-row M tile, two 16-col U subtiles, 4 gates = 8 v8f accumulators.
// XK: 0 = x source is f32 (layer 1), 1 = x source is bf16 (layers 2/3).
// ---------------------------------------------------------------------------
template<int XK>
__global__ __launch_bounds__(256) void lstm_gemm_cell_kernel(
    const float*          __restrict__ a_x_f32,
    const unsigned short* __restrict__ a_x_bf16,
    int                   x_row_stride,
    int                   Kx,
    const unsigned short* __restrict__ h_in,
    const unsigned short* __restrict__ wt,        // [4096][Ktot] bf16 col-major
    int                   Ktot,
    const float*          __restrict__ bias,      // 4096
    float*                __restrict__ c_state,   // B x 1024 f32, in-place
    unsigned short*       __restrict__ h_out)     // B x 1024 bf16 (ping-pong)
{
    __shared__ unsigned short lds_a[2][64 * 32];      // 2 x 4 KB
    __shared__ unsigned short lds_b[2][4 * 64 * 32];  // 2 x 16 KB

    const int tid   = threadIdx.x;
    const int lane  = tid & 31;
    const int wave  = tid >> 5;
    const int wm    = wave >> 1;        // 0..3  (M position, 16 rows)
    const int wu    = wave & 1;         // 0..1  (U position, 32 cols)
    const int half  = lane >> 4;
    const int lr    = lane & 15;

    const int mbase = blockIdx.y * 64;  // batch row base
    const int ubase = blockIdx.x * 64;  // u base

    const int srow = tid >> 2;          // staging row/col 0..63
    const int sseg = tid & 3;           // 16B segment (8 bf16)

    // Hoisted per-thread base pointers (k added per chunk)
    const float*          xrow_f = a_x_f32  + (size_t)(mbase + srow) * x_row_stride;
    const unsigned short* xrow_h = a_x_bf16 + (size_t)(mbase + srow) * x_row_stride;
    const unsigned short* hrow   = h_in     + (size_t)(mbase + srow) * 1024 - Kx;
    const unsigned short* wrow   = wt + (size_t)(ubase + srow) * Ktot;
    const size_t          wgate  = (size_t)1024 * Ktot;   // gate stride in wt

    // Per-lane LDS fragment offsets (ushort units), per ISA 7.12.2 layouts
    const int aoff = (wm * 16 + lr) * 32 + half * 8;          // + {0,16}
    const int boff = (wu * 32 + lr) * 32 + half * 8;          // + st*512 + g*2048 + {0,16}

    // Issue the staging transfers for K-chunk kc into LDS buffer pb.
    // 4 or 5 transfers per thread; async ones retire in order.
    auto issue_tile = [&](int kc, int pb) {
        const int kg = (kc << 5) + sseg * 8;
        unsigned short* adst = &lds_a[pb][srow * 32 + sseg * 8];
        if ((kc << 5) < Kx) {                      // uniform (Kx % 32 == 0)
            if (XK == 0) {
                // f32 -> bf16 must go through VALU: manual staging (layer-1
                // x-phase only: 8 of 40 chunks).
                const float* s = xrow_f + kg;
                flt4 p0 = *(const flt4*)(s);
                flt4 p1 = *(const flt4*)(s + 4);
                unsigned short t[8];
                t[0] = f2bf(p0.x); t[1] = f2bf(p0.y);
                t[2] = f2bf(p0.z); t[3] = f2bf(p0.w);
                t[4] = f2bf(p1.x); t[5] = f2bf(p1.y);
                t[6] = f2bf(p1.z); t[7] = f2bf(p1.w);
                *(ui4*)adst = *(ui4*)t;
            } else {
                cp16_g2l(xrow_h + kg, adst);
            }
        } else {
            cp16_g2l(hrow + kg, adst);             // hrow already biased by -Kx
        }
        #pragma unroll
        for (int g = 0; g < 4; ++g)
            cp16_g2l(wrow + (size_t)g * wgate + kg,
                     &lds_b[pb][(g * 64 + srow) * 32 + sseg * 8]);
    };

    v8f acc[2][4] = {};                 // [u-subtile][gate i,f,g,o]

    const int nk = Ktot >> 5;
    issue_tile(0, 0);
    int p = 0;
    for (int kc = 0; kc < nk; ++kc) {
        // Barrier A: all waves finished reading buffer p^1 (chunk kc-1), so
        // it may be overwritten by chunk kc+1's DMA.
        __syncthreads();

        if (kc + 1 < nk) {
            issue_tile(kc + 1, p ^ 1);     // DMA in flight during WMMA below
            // Asyncs complete in order: leaving <= (next tile's count)
            // outstanding guarantees chunk kc's transfers have landed.
            // XK==0 tiles issue only 4 asyncs (A staged manually), so the
            // conservative bound there is 4.
            if (XK == 0) wait_async<4>(); else wait_async<5>();
        } else {
            wait_async<0>();
        }

        // Barrier B: buffer p contents visible to every wave.
        __syncthreads();

        // ---- fragments: load all, then WMMA back-to-back ------------------
        v16bf af;
        ((ui4*)&af)[0] = *(ui4*)&lds_a[p][aoff];
        ((ui4*)&af)[1] = *(ui4*)&lds_a[p][aoff + 16];

        v16bf bfm[2][4];
        #pragma unroll
        for (int st = 0; st < 2; ++st)
            #pragma unroll
            for (int g = 0; g < 4; ++g) {
                const int co = boff + g * 2048 + st * 512;
                ((ui4*)&bfm[st][g])[0] = *(ui4*)&lds_b[p][co];
                ((ui4*)&bfm[st][g])[1] = *(ui4*)&lds_b[p][co + 16];
            }

        #pragma unroll
        for (int st = 0; st < 2; ++st)
            #pragma unroll
            for (int g = 0; g < 4; ++g)
                acc[st][g] = __builtin_amdgcn_wmma_f32_16x16x32_bf16(
                    false, af, false, bfm[st][g], (short)0, acc[st][g],
                    false, false);

        p ^= 1;
    }

    // ---- fused LSTM cell epilogue on accumulator registers ----------------
    // C layout: VGPR r, lane l ->  M = r + 8*(l>=16),  N = l%16
    const int mrow0 = mbase + wm * 16 + half * 8;
    #pragma unroll
    for (int st = 0; st < 2; ++st) {
        const int u = ubase + wu * 32 + st * 16 + lr;
        const float bi  = bias[u];
        const float bf_ = bias[1024 + u];
        const float bg  = bias[2048 + u];
        const float bo  = bias[3072 + u];
        #pragma unroll
        for (int r = 0; r < 8; ++r) {
            const size_t idx = (size_t)(mrow0 + r) * 1024 + u;
            const float ig = fsigmoid(acc[st][0][r] + bi);
            const float fg = fsigmoid(acc[st][1][r] + bf_);
            const float gg = ftanh   (acc[st][2][r] + bg);
            const float og = fsigmoid(acc[st][3][r] + bo);
            const float cn = fg * c_state[idx] + ig * gg;
            c_state[idx] = cn;
            h_out[idx]   = f2bf(og * ftanh(cn));
        }
    }
}

// ---------------------------------------------------------------------------
// Heads: out = (c3@w_mean + b_mean) + exp((c3@w_sigma + b_sigma)/2) + eps
// ---------------------------------------------------------------------------
__global__ __launch_bounds__(128) void heads_kernel(
    const float* __restrict__ c3,
    const float* __restrict__ wmean, const float* __restrict__ bmean,
    const float* __restrict__ wsig,  const float* __restrict__ bsig,
    const float* __restrict__ eps,   float* __restrict__ out)
{
    __shared__ float crow[1024];
    const int b = blockIdx.x;
    const int l = threadIdx.x;
    for (int i = l; i < 1024; i += 128) crow[i] = c3[(size_t)b * 1024 + i];
    __syncthreads();
    float m = bmean[l], s = bsig[l];
    #pragma unroll 4
    for (int u = 0; u < 1024; ++u) {
        const float cv = crow[u];
        m += cv * wmean[(size_t)u * 128 + l];
        s += cv * wsig [(size_t)u * 128 + l];
    }
    out[(size_t)b * 128 + l] = m + __expf(0.5f * s) + eps[(size_t)b * 128 + l];
}

// ---------------------------------------------------------------------------
extern "C" void kernel_launch(void* const* d_in, const int* in_sizes, int n_in,
                              void* d_out, int out_size, void* d_ws, size_t ws_size,
                              hipStream_t stream)
{
    (void)in_sizes; (void)n_in; (void)out_size; (void)ws_size;

    const float* x      = (const float*)d_in[0];
    const float* W1     = (const float*)d_in[1];
    const float* U1     = (const float*)d_in[2];
    const float* b1     = (const float*)d_in[3];
    const float* W2     = (const float*)d_in[4];
    const float* U2     = (const float*)d_in[5];
    const float* b2     = (const float*)d_in[6];
    const float* W3     = (const float*)d_in[7];
    const float* U3     = (const float*)d_in[8];
    const float* b3     = (const float*)d_in[9];
    const float* w_mean = (const float*)d_in[10];
    const float* b_mean = (const float*)d_in[11];
    const float* w_sig  = (const float*)d_in[12];
    const float* b_sig  = (const float*)d_in[13];
    const float* eps    = (const float*)d_in[14];
    float* out          = (float*)d_out;

    // Workspace layout (~48 MB total)
    char* w = (char*)d_ws;
    unsigned short* wt1 = (unsigned short*)(w);                       // 1280*4096*2
    unsigned short* wt2 = (unsigned short*)(w + 10485760);            // 2048*4096*2
    unsigned short* wt3 = (unsigned short*)(w + 27262976);            // 2048*4096*2
    unsigned short* hb  = (unsigned short*)(w + 44040192);            // 6 * 256*1024 bf16
    float*          cb  = (float*)(w + 47185920);                     // 3 * 256*1024 f32
    unsigned short* h[3][2];
    float* c[3];
    for (int l = 0; l < 3; ++l) {
        h[l][0] = hb + (size_t)(2 * l + 0) * 256 * 1024;
        h[l][1] = hb + (size_t)(2 * l + 1) * 256 * 1024;
        c[l]    = cb + (size_t)l * 256 * 1024;
    }

    // 1) bf16 + transpose weights (once per launch; L2-resident afterwards)
    transpose_cat_kernel<<<1280 * 16, 256, 0, stream>>>(W1, U1,  256, 1280, wt1);
    transpose_cat_kernel<<<2048 * 16, 256, 0, stream>>>(W2, U2, 1024, 2048, wt2);
    transpose_cat_kernel<<<2048 * 16, 256, 0, stream>>>(W3, U3, 1024, 2048, wt3);

    // 2) zero h (ping 0) and c
    init_state_kernel<<<1024, 256, 0, stream>>>(h[0][0], h[1][0], h[2][0],
                                                c[0], c[1], c[2]);

    // 3) sequential scan: 256 steps x 3 layers of fused GEMM+cell
    const dim3 grid(16, 4);   // 16 u-tiles x 4 batch-tiles
    for (int t = 0; t < 256; ++t) {
        const int pi = t & 1, po = pi ^ 1;
        // layer 1: x_t (f32, row stride T*F = 65536), K = 256 + 1024
        lstm_gemm_cell_kernel<0><<<grid, 256, 0, stream>>>(
            x + (size_t)t * 256, nullptr, 65536, 256,
            h[0][pi], wt1, 1280, b1, c[0], h[0][po]);
        // layer 2: x = h1_out (bf16), K = 1024 + 1024
        lstm_gemm_cell_kernel<1><<<grid, 256, 0, stream>>>(
            nullptr, h[0][po], 1024, 1024,
            h[1][pi], wt2, 2048, b2, c[1], h[1][po]);
        // layer 3
        lstm_gemm_cell_kernel<1><<<grid, 256, 0, stream>>>(
            nullptr, h[1][po], 1024, 1024,
            h[2][pi], wt3, 2048, b3, c[2], h[2][po]);
    }

    // 4) heads off final cell state c3
    heads_kernel<<<256, 128, 0, stream>>>(c[2], w_mean, b_mean, w_sig, b_sig,
                                          eps, out);
}
